// LinearAttention_70007966925361
// MI455X (gfx1250) — matrix operand order repaired and checked by reading, not compiled, not run on hardware
//
#include <hip/hip_runtime.h>
#include <hip/hip_bf16.h>

typedef __attribute__((ext_vector_type(8)))  float   v8f;
typedef __attribute__((ext_vector_type(16))) __bf16  v16bf;
typedef __attribute__((ext_vector_type(8)))  __bf16  v8bf;

// ---- CDNA5 async global->LDS copy via inline asm (builtin's pointer params
// use HIP language address spaces that aren't spellable from source).
// Set to 0 to fall back to plain vector copies through VGPRs.
#define USE_ASYNC_ASM 1

__device__ __forceinline__ void async_copy16(const void* gsrc, void* ldst) {
#if USE_ASYNC_ASM
    // VDST = per-lane LDS byte address (low 32 bits of generic LDS pointer),
    // VADDR = 64-bit global address, GV mode (no SADDR). Tracked by ASYNCcnt.
    asm volatile("global_load_async_to_lds_b128 %0, %1, off"
                 :
                 : "v"((unsigned)(uintptr_t)ldst), "v"(gsrc)
                 : "memory");
#else
    *(uint4*)ldst = *(const uint4*)gsrc;
#endif
}

__device__ __forceinline__ void wait_async0() {
#if USE_ASYNC_ASM
    asm volatile("s_wait_asynccnt 0" ::: "memory");
#endif
}

// ---------------------------------------------------------------------------
static constexpr long long BB  = 2;
static constexpr long long CC  = 256;
static constexpr long long NN  = 16384;   // T*H*W
static constexpr long long HID = 512;     // 8 heads * 64

__device__ __forceinline__ v16bf cat16(v8bf lo, v8bf hi) {
    return __builtin_shufflevector(lo, hi, 0,1,2,3,4,5,6,7,8,9,10,11,12,13,14,15);
}

__global__ void cvt_f32_bf16(const float* __restrict__ in, __bf16* __restrict__ out,
                             long long n) {
    long long i = (long long)blockIdx.x * blockDim.x + threadIdx.x;
    if (i < n) out[i] = (__bf16)in[i];
}

__global__ void zero_f32(float* __restrict__ p, int n) {
    int i = blockIdx.x * blockDim.x + threadIdx.x;
    if (i < n) p[i] = 0.0f;
}

// ---------------------------------------------------------------------------
// bf16 WMMA GEMM:  Out[m,n] = sum_k W[m,k] * X[k,n]
//  * 256 threads = 8 waves (2x4), K stepped by 32, LDS double-buffered
//  * A tile staged row-major via async global->LDS b128 copies (ASYNCcnt)
//  * B tile staged as K-pair-packed dwords: lds_b[n][kp] = (X[2kp,n], X[2kp+1,n])
//    -> each lane's B fragment is two contiguous 16B LDS reads
//  * LDS row strides 80B: 16B aligned, conflict-free fragment reads
// ---------------------------------------------------------------------------
template <int TM, int TN, bool OUT_BF16, bool ADD_BIAS>
__global__ __launch_bounds__(256) void gemm_wmma(
    const __bf16* __restrict__ Wm, const __bf16* __restrict__ Xm,
    void* __restrict__ Om, const float* __restrict__ bias,
    int K, int N, int ldW,
    long long sW1, long long sW2, long long sX1, long long sX2,
    long long sO1, long long sO2, int zdiv)
{
    constexpr int LDA  = 40;   // bf16 elems per A row (80B)
    constexpr int LDBW = 20;   // dwords per B row (80B)
    __shared__ __bf16   lds_a[2][TM * LDA];
    __shared__ unsigned lds_b[2][TN * LDBW];

    const int z = blockIdx.z;
    const __bf16* Wp = Wm + (long long)z * sW1 + (long long)(z / zdiv) * sW2;
    const __bf16* Xp = Xm + (long long)z * sX1 + (long long)(z / zdiv) * sX2;
    const long long oofs = (long long)z * sO1 + (long long)(z / zdiv) * sO2;

    const int n0  = blockIdx.x * TN;
    const int m0  = blockIdx.y * TM;
    const int tid = threadIdx.x;
    const int lane = tid & 31, wid = tid >> 5;

    constexpr int WROWS = 2, WCOLS = 4;
    constexpr int WM = TM / WROWS, WN = TN / WCOLS;
    constexpr int MT = WM / 16,    NT = WN / 16;
    const int wm = (wid / WCOLS) * WM;
    const int wn = (wid % WCOLS) * WN;

    const int l15 = lane & 15;
    const int kbA = (lane < 16) ? 0 : 8;   // A frag K-phase (elems)
    const int kbB = (lane < 16) ? 0 : 4;   // B frag K-phase (dwords = K pairs)

    auto stageA = [&](int buf, int k0) {
        for (int q = tid; q < TM * 4; q += 256) {
            int row = q >> 2, cc = (q & 3) << 3;
            async_copy16(Wp + (long long)(m0 + row) * ldW + k0 + cc,
                         &lds_a[buf][row * LDA + cc]);
        }
    };
    auto stageB = [&](int buf, int k0) {
        for (int q = tid; q < (TN / 8) * 16; q += 256) {
            int kp = q / (TN / 8);          // K-pair index 0..15
            int c8 = (q % (TN / 8)) << 3;   // n chunk
            const __bf16* s0 = Xp + (long long)(k0 + 2 * kp) * N + n0 + c8;
            v8bf lo = *(const v8bf*)s0;
            v8bf hi = *(const v8bf*)(s0 + N);
#pragma unroll
            for (int j = 0; j < 8; ++j) {
                union { __bf16 h[2]; unsigned u; } pk;
                pk.h[0] = lo[j]; pk.h[1] = hi[j];
                lds_b[buf][(c8 + j) * LDBW + kp] = pk.u;
            }
        }
    };

    v8f acc[MT][NT] = {};

    stageA(0, 0); stageB(0, 0);
    wait_async0();
    __syncthreads();

    int buf = 0;
    for (int k0 = 0; k0 < K; k0 += 32) {
        if (k0 + 32 < K) { stageA(buf ^ 1, k0 + 32); stageB(buf ^ 1, k0 + 32); }

        v16bf af[MT], bfm[NT];
#pragma unroll
        for (int i = 0; i < MT; ++i) {
            const __bf16* p = &lds_a[buf][(wm + i * 16 + l15) * LDA + kbA];
            af[i] = cat16(*(const v8bf*)p, *(const v8bf*)(p + 16));
        }
#pragma unroll
        for (int j = 0; j < NT; ++j) {
            const unsigned* p = &lds_b[buf][(wn + j * 16 + l15) * LDBW + kbB];
            bfm[j] = cat16(*(const v8bf*)p, *(const v8bf*)(p + 8));
        }
#pragma unroll
        for (int i = 0; i < MT; ++i)
#pragma unroll
            for (int j = 0; j < NT; ++j)
                acc[i][j] = __builtin_amdgcn_wmma_f32_16x16x32_bf16(
                    false, af[i], false, bfm[j], (short)0, acc[i][j], false, false);

        wait_async0();
        __syncthreads();
        buf ^= 1;
    }

    // D layout: VGPR r, lane -> row = r + 8*(lane>=16), col = lane&15
    const int lh8 = (lane >> 4) * 8;
#pragma unroll
    for (int i = 0; i < MT; ++i)
#pragma unroll
        for (int j = 0; j < NT; ++j) {
            const int col = n0 + wn + j * 16 + l15;
#pragma unroll
            for (int r = 0; r < 8; ++r) {
                const int row = m0 + wm + i * 16 + lh8 + r;
                float val = acc[i][j][r];
                if (ADD_BIAS) val += bias[row];
                if (OUT_BF16)
                    ((__bf16*)Om)[oofs + (long long)row * N + col] = (__bf16)val;
                else
                    ((float*)Om)[oofs + (long long)row * N + col] = val;
            }
        }
}

// ---------------------------------------------------------------------------
// Row softmax over N=16384 (in place on the k slice of qkv; f32 math)
// ---------------------------------------------------------------------------
__global__ __launch_bounds__(256) void softmax_rows(__bf16* __restrict__ qkv, int N) {
    __shared__ float red[256];
    const int rb = blockIdx.x;             // b*512 + row
    const long long b = rb >> 9;
    const int r = rb & 511;
    __bf16* p = qkv + (b * 1536 + 512 + r) * NN;
    const int tid = threadIdx.x;

    float m = -3.4e38f;
    for (int i = tid; i < N; i += 256) m = fmaxf(m, (float)p[i]);
    red[tid] = m; __syncthreads();
    for (int s = 128; s; s >>= 1) { if (tid < s) red[tid] = fmaxf(red[tid], red[tid + s]); __syncthreads(); }
    const float M = red[0]; __syncthreads();

    float sum = 0.0f;
    for (int i = tid; i < N; i += 256) sum += __expf((float)p[i] - M);
    red[tid] = sum; __syncthreads();
    for (int s = 128; s; s >>= 1) { if (tid < s) red[tid] += red[tid + s]; __syncthreads(); }
    const float inv = 1.0f / red[0]; __syncthreads();

    for (int i = tid; i < N; i += 256) p[i] = (__bf16)(__expf((float)p[i] - M) * inv);
}

// ---------------------------------------------------------------------------
// context[b,h,d,e] = sum_n k_sm[d,n] * v[e,n].  K (=n) contiguous for both
// operands -> fragments straight from global; split-K + f32 atomics.
// ---------------------------------------------------------------------------
__global__ __launch_bounds__(128) void ctx_wmma(
    const __bf16* __restrict__ qkv, float* __restrict__ ctx, int N, int klen)
{
    const int bh = blockIdx.x & 15;
    const int kc = blockIdx.x >> 4;
    const long long b = bh >> 3, h = bh & 7;
    const __bf16* kp = qkv + (b * 1536 + 512  + h * 64) * NN;
    const __bf16* vp = qkv + (b * 1536 + 1024 + h * 64) * NN;

    const int lane = threadIdx.x & 31, wid = threadIdx.x >> 5;
    const int l15 = lane & 15;
    const int kbA = (lane < 16) ? 0 : 8;
    const int kbB = (lane < 16) ? 0 : 8;

    v8f acc[4] = {};
    const int kstart = kc * klen;
    for (int n0 = kstart; n0 < kstart + klen; n0 += 32) {
        const __bf16* pa = kp + (long long)(wid * 16 + l15) * N + n0 + kbA;
        v16bf a = cat16(*(const v8bf*)pa, *(const v8bf*)(pa + 16));
#pragma unroll
        for (int t = 0; t < 4; ++t) {
            const __bf16* pb = vp + (long long)(t * 16 + l15) * N + n0 + kbB;
            v16bf bb = cat16(*(const v8bf*)pb, *(const v8bf*)(pb + 16));
            acc[t] = __builtin_amdgcn_wmma_f32_16x16x32_bf16(
                false, a, false, bb, (short)0, acc[t], false, false);
        }
    }
    float* cbase = ctx + (long long)bh * 4096;
    const int lh8 = (lane >> 4) * 8;
#pragma unroll
    for (int t = 0; t < 4; ++t)
#pragma unroll
        for (int r = 0; r < 8; ++r)
            atomicAdd(&cbase[(wid * 16 + lh8 + r) * 64 + t * 16 + l15], acc[t][r]);
}

// ctxT[bh][e][d] = bf16(ctx[bh][d][e])
__global__ void ctx_transpose(const float* __restrict__ ctx, __bf16* __restrict__ ctxT) {
    int i = blockIdx.x * 256 + threadIdx.x;      // 65536
    int bh = i >> 12, d = (i >> 6) & 63, e = i & 63;
    ctxT[(bh << 12) + (e << 6) + d] = (__bf16)ctx[i];
}

// ---------------------------------------------------------------------------
extern "C" void kernel_launch(void* const* d_in, const int* in_sizes, int n_in,
                              void* d_out, int out_size, void* d_ws, size_t ws_size,
                              hipStream_t stream) {
    const float* x      = (const float*)d_in[0];
    const float* w_qkv  = (const float*)d_in[1];
    const float* w_out  = (const float*)d_in[2];
    const float* b_out  = (const float*)d_in[3];
    float*       out    = (float*)d_out;

    char* ws = (char*)d_ws;
    auto alloc = [&](size_t bytes) -> char* {
        char* p = ws; ws += (bytes + 255) & ~(size_t)255; return p;
    };
    __bf16* xb    = (__bf16*)alloc(BB * CC * NN * 2);
    __bf16* wqkvb = (__bf16*)alloc(3 * HID * CC * 2);
    __bf16* woutb = (__bf16*)alloc(CC * HID * 2);
    __bf16* qkvb  = (__bf16*)alloc(BB * 3 * HID * NN * 2);
    __bf16* out2b = (__bf16*)alloc(BB * HID * NN * 2);
    float*  ctxf  = (float*)alloc(16 * 4096 * 4);
    __bf16* ctxTb = (__bf16*)alloc(16 * 4096 * 2);

    // 1) convert to bf16
    {
        long long n = BB * CC * NN;
        cvt_f32_bf16<<<(int)((n + 255) / 256), 256, 0, stream>>>(x, xb, n);
        n = 3 * HID * CC;
        cvt_f32_bf16<<<(int)((n + 255) / 256), 256, 0, stream>>>(w_qkv, wqkvb, n);
        n = CC * HID;
        cvt_f32_bf16<<<(int)((n + 255) / 256), 256, 0, stream>>>(w_out, woutb, n);
    }

    // 2) qkv = w_qkv @ x   (M=1536, K=256, N=16384, batched over B)
    gemm_wmma<128, 128, true, false><<<dim3(128, 12, 2), 256, 0, stream>>>(
        wqkvb, xb, qkvb, nullptr, 256, 16384, 256,
        0, 0, CC * NN, 0, 3 * HID * NN, 0, 8);

    // 3) k = softmax(k) over N (in place)
    softmax_rows<<<1024, 256, 0, stream>>>(qkvb, 16384);

    // 4) context = k @ v^T   (split-K atomics)
    zero_f32<<<256, 256, 0, stream>>>(ctxf, 16 * 4096);
    ctx_wmma<<<16 * 16, 128, 0, stream>>>(qkvb, ctxf, 16384, 1024);
    ctx_transpose<<<256, 256, 0, stream>>>(ctxf, ctxTb);

    // 5) out2 = ctx^T @ q   (M=64, K=64, N=16384, z = b*8+h)
    gemm_wmma<64, 128, true, false><<<dim3(128, 1, 16), 256, 0, stream>>>(
        ctxTb, qkvb, out2b, nullptr, 64, 16384, 64,
        4096, 0, 64 * NN, 1024 * NN, 64 * NN, 0, 8);

    // 6) out = w_out @ out2 + b_out   (M=256, K=512, N=16384, f32 out)
    gemm_wmma<128, 128, false, true><<<dim3(128, 2, 2), 256, 0, stream>>>(
        woutb, out2b, out, b_out, 512, 16384, 512,
        0, 0, HID * NN, 0, CC * NN, 0, 8);
}
